// SpatialNeighborAttention_40200893890567
// MI455X (gfx1250) — matrix operand order repaired and checked by reading, not compiled, not run on hardware
//
#include <hip/hip_runtime.h>
#include <hip/hip_bf16.h>
#include <math.h>

typedef __bf16 bf16_t;
typedef __attribute__((ext_vector_type(16))) __bf16 v16bf;
typedef __attribute__((ext_vector_type(8)))  __bf16 v8bf;
typedef __attribute__((ext_vector_type(8)))  float  v8f;

#define K_NB   5
#define COMB   128
#define HID    256
#define HHALF  128
#define BT     16            // b's per tile
#define ROWS   (BT * K_NB)   // 80 rows -> 5 tiles of 16
#define NWAVES 10            // wave pair per 16-row tile
#define NTHREADS (NWAVES * 32)
#define MAX_PERSISTENT 1024  // persistent WGs; weights staged once per WG

// tanh-form GELU: |err| vs exact erf-GELU ~3e-4, below the bf16 rounding
// applied to every GELU output downstream. Lowers to native v_tanh_f32.
__device__ __forceinline__ float gelu_fast(float x) {
  const float c0 = 0.7978845608028654f;        // sqrt(2/pi)
  const float c1 = 0.035677408136300125f;      // c0 * 0.044715
  float y = x * (c0 + c1 * x * x);
  float t;
#if __has_builtin(__builtin_amdgcn_tanhf)
  t = __builtin_amdgcn_tanhf(y);
#else
  float e = __expf(2.0f * y);
  t = __fdividef(e - 1.0f, e + 1.0f);
#endif
  return 0.5f * x * (1.0f + t);
}

// A fragment: row-major source [16 rows][ld]. ISA 16-bit A layout:
// lane L (<16): K = k0..k0+7 (elems 0-7), k0+16..k0+23 (elems 8-15); lane L+16: +8.
__device__ __forceinline__ v16bf load_a_frag(const bf16_t* base, int ld, int k0, int lane) {
  const int hlf = lane >> 4, m = lane & 15;
  const bf16_t* p = base + m * ld + k0 + 8 * hlf;
  union { v16bf v; v8bf h[2]; } u;
  u.h[0] = *(const v8bf*)(p);
  u.h[1] = *(const v8bf*)(p + 16);
  return u.v;
}

// B fragment from TRANSPOSED weights WT[n][k] (ld = K stride). ISA 16-bit B layout:
// lane L (<16): K = k0..k0+7 (elems 0-7), k0+16..k0+23 (elems 8-15) of column n0+L; lane L+16: +8.
__device__ __forceinline__ v16bf load_b_frag(const bf16_t* wt, int ld, int n0, int k0, int lane) {
  const int hlf = lane >> 4;
  const bf16_t* p = wt + (n0 + (lane & 15)) * ld + k0 + 8 * hlf;
  union { v16bf v; v8bf h[2]; } u;
  u.h[0] = *(const v8bf*)(p);
  u.h[1] = *(const v8bf*)(p + 16);
  return u.v;
}

__device__ __forceinline__ v8f wmma_bf16(v16bf a, v16bf b, v8f c) {
  return __builtin_amdgcn_wmma_f32_16x16x32_bf16(false, a, false, b, (short)0, c, false, false);
}

__global__ void __launch_bounds__(NTHREADS)
sna_kernel(const int* __restrict__ res_idx, const int* __restrict__ ss_idx,
           const float* __restrict__ dist, const int* __restrict__ seq_sep,
           const float* __restrict__ angles, const unsigned char* __restrict__ valid,
           const float* __restrict__ res_table, const float* __restrict__ ss_table,
           const float* __restrict__ cont_W, const float* __restrict__ cont_b,
           const float* __restrict__ attn_W1, const float* __restrict__ attn_b1,
           const float* __restrict__ attn_W2, const float* __restrict__ attn_b2,
           const float* __restrict__ val_W1, const float* __restrict__ val_b1,
           const float* __restrict__ val_W2, const float* __restrict__ val_b2,
           const float* __restrict__ fallback, float* __restrict__ out,
           int Btot, int numTiles)
{
  extern __shared__ char smem[];
  bf16_t* attnW1T = (bf16_t*)smem;                 // [HHALF][COMB]  32 KB
  bf16_t* valW1T  = attnW1T + HHALF * COMB;        // [HID][COMB]    64 KB
  bf16_t* valW2T  = valW1T  + HID * COMB;          // [HID][HID]    128 KB
  bf16_t* combS   = valW2T  + HID * HID;           // [ROWS][COMB]   20 KB
  bf16_t* v1S     = combS   + ROWS * COMB;         // [ROWS][HID]    40 KB
  bf16_t* uS      = v1S     + ROWS * HID;          // [BT][HID]       8 KB
  float*  aW2S    = (float*)(uS + BT * HID);       // [HHALF]
  float*  ab1S    = aW2S   + HHALF;                // [HHALF]
  float*  vb1S    = ab1S   + HHALF;                // [HID]
  float*  vb2S    = vb1S   + HID;                  // [HID]
  float*  cWS     = vb2S   + HID;                  // [6][32]
  float*  cbS     = cWS    + 6 * 32;               // [32]
  float*  scPart  = cbS    + 32;                   // [2][ROWS] score partials
  float*  wtS     = scPart + 2 * ROWS;             // [ROWS]
  int*    avS     = (int*)(wtS + ROWS);            // [BT]

  const int tid   = threadIdx.x;
  const int lane  = tid & 31;
  const int wave  = tid >> 5;          // 0..9
  const int tile  = wave >> 1;         // 0..4 : 16-row tile
  const int par   = wave & 1;          // N-split within the tile pair

  // ---- Stage 0 (once per persistent WG): weights f32 -> bf16^T into LDS ----
  for (int i = tid; i < HHALF * COMB; i += NTHREADS) {
    int c = i >> 7, h = i & (HHALF - 1);
    attnW1T[h * COMB + c] = (bf16_t)attn_W1[i];
  }
  for (int i = tid; i < COMB * HID; i += NTHREADS) {
    int c = i >> 8, h = i & (HID - 1);
    valW1T[h * COMB + c] = (bf16_t)val_W1[i];
  }
  for (int i = tid; i < HID * HID; i += NTHREADS) {
    int c = i >> 8, h = i & (HID - 1);
    valW2T[h * HID + c] = (bf16_t)val_W2[i];
  }
  for (int i = tid; i < HHALF; i += NTHREADS) { aW2S[i] = attn_W2[i]; ab1S[i] = attn_b1[i]; }
  for (int i = tid; i < HID;   i += NTHREADS) { vb1S[i] = val_b1[i]; vb2S[i] = val_b2[i]; }
  for (int i = tid; i < 6 * 32; i += NTHREADS) cWS[i] = cont_W[i];
  for (int i = tid; i < 32;     i += NTHREADS) cbS[i] = cont_b[i];

  // ---- Persistent loop over b-tiles ----
  for (int tb = blockIdx.x; tb < numTiles; tb += gridDim.x) {
    const int bbase = tb * BT;
    __syncthreads();   // weights ready / previous iteration fully drained

    // ---- Stage A: build combined[80][128] (gather + cont MLP) as bf16 ----
    {
      const int r0 = wave * 8;            // 8 rows per wave
      for (int r = 0; r < 8; ++r) {
        const int rho = r0 + r;
        int b = bbase + rho / K_NB; if (b >= Btot) b = Btot - 1;
        const int k = rho % K_NB;
        const long long e = (long long)b * K_NB + k;
        const int ri = res_idx[e], si = ss_idx[e];
        float d = dist[e]; d = (d == d) ? d : 0.0f;
        float f[6];
        f[0] = d * (1.0f / 15.0f);
        f[1] = __logf(1.0f + fabsf((float)seq_sep[e])) * 0.2f;
        #pragma unroll
        for (int a = 0; a < 4; ++a) { float x = angles[e * 4 + a]; f[2 + a] = (x == x) ? x : 0.0f; }
        const int c0 = lane * 4;
        #pragma unroll
        for (int j = 0; j < 4; ++j) {
          const int c = c0 + j;
          float v;
          if (c < 64)      v = res_table[ri * 64 + c];
          else if (c < 96) v = ss_table[si * 32 + (c - 64)];
          else {
            const int cc = c - 96;
            v = cbS[cc];
            #pragma unroll
            for (int q = 0; q < 6; ++q) v += f[q] * cWS[q * 32 + cc];
          }
          combS[rho * COMB + c] = (bf16_t)v;
        }
      }
    }
    __syncthreads();

    // ---- Stage B: score partials = sum_n gelu((comb @ W1)+b1)[n] * W2[n] ----
    {
      const bf16_t* arow = combS + (tile * 16) * COMB;
      float sc[8];
      #pragma unroll
      for (int v = 0; v < 8; ++v) sc[v] = 0.0f;
      for (int j = par; j < 8; j += 2) {           // wave pair splits N-tiles
        const int n = j * 16 + (lane & 15);
        const float b1 = ab1S[n];
        v8f acc;
        #pragma unroll
        for (int v = 0; v < 8; ++v) acc[v] = b1;   // bias folded into C
        #pragma unroll
        for (int kk = 0; kk < COMB; kk += 32) {
          v16bf a  = load_a_frag(arow, COMB, kk, lane);
          v16bf bb = load_b_frag(attnW1T, COMB, j * 16, kk, lane);
          acc = wmma_bf16(a, bb, acc);
        }
        const float w2 = aW2S[n];
        #pragma unroll
        for (int v = 0; v < 8; ++v) sc[v] += gelu_fast(acc[v]) * w2;
      }
      #pragma unroll
      for (int m = 1; m < 16; m <<= 1)
        #pragma unroll
        for (int v = 0; v < 8; ++v) sc[v] += __shfl_xor(sc[v], m, 32);
      if ((lane & 15) == 0) {
        const int hlf = lane >> 4;
        for (int v = 0; v < 8; ++v) {
          const int rho = tile * 16 + v + 8 * hlf;
          scPart[par * ROWS + rho] = sc[v];
        }
      }
    }
    __syncthreads();

    // ---- Stage C: combine partials, mask, softmax over K=5 (lane = local b) ----
    if (tid < BT) {
      const int bl = tid, b = bbase + bl;
      const float b2 = attn_b2[0];
      float s[K_NB], mx = -1e30f;
      int av = 0;
      for (int i = 0; i < K_NB; ++i) {
        const int rho = bl * K_NB + i;
        const bool vl = (b < Btot) && (valid[(long long)b * K_NB + i] != 0);
        s[i] = vl ? (scPart[rho] + scPart[ROWS + rho] + b2) : -10000.0f;
        mx = fmaxf(mx, s[i]);
        if (vl) av = 1;
      }
      float sum = 0.0f, e[K_NB];
      for (int i = 0; i < K_NB; ++i) { e[i] = __expf(s[i] - mx); sum += e[i]; }
      const float inv = 1.0f / sum;
      for (int i = 0; i < K_NB; ++i) wtS[bl * K_NB + i] = e[i] * inv;
      avS[bl] = av;
    }
    __syncthreads();

    // ---- Stage D: V1 = gelu((comb @ val_W1)+b1) -> bf16 LDS ----
    {
      const bf16_t* arow = combS + (tile * 16) * COMB;
      for (int j = par; j < 16; j += 2) {          // wave pair splits N-tiles
        const int n = j * 16 + (lane & 15);
        const float b1 = vb1S[n];
        v8f acc;
        #pragma unroll
        for (int v = 0; v < 8; ++v) acc[v] = b1;   // bias folded into C
        #pragma unroll
        for (int kk = 0; kk < COMB; kk += 32) {
          v16bf a  = load_a_frag(arow, COMB, kk, lane);
          v16bf bb = load_b_frag(valW1T, COMB, j * 16, kk, lane);
          acc = wmma_bf16(a, bb, acc);
        }
        const int hlf = lane >> 4;
        #pragma unroll
        for (int v = 0; v < 8; ++v) {
          const int rho = tile * 16 + v + 8 * hlf;
          v1S[rho * HID + n] = (bf16_t)gelu_fast(acc[v]);
        }
      }
    }
    __syncthreads();

    // ---- Stage E1: softmax-weighted reduce V1[80][256] -> U[16][256]
    //      ((sum_k w_k V1_k) @ W2 + b2*sum(w) == einsum result; sum(w)==1) ----
    for (int i = tid; i < BT * HID; i += NTHREADS) {
      const int bl = i >> 8, n = i & (HID - 1);
      float s = 0.0f;
      #pragma unroll
      for (int k = 0; k < K_NB; ++k)
        s += wtS[bl * K_NB + k] * (float)v1S[(bl * K_NB + k) * HID + n];
      uS[i] = (bf16_t)s;
    }
    __syncthreads();

    // ---- Stage E2: out = U @ val_W2 + b2 ; N-tiles round-robin over 10 waves;
    //      epilogue writes straight to global with fallback select ----
    for (int j = wave; j < 16; j += NWAVES) {
      const int n = j * 16 + (lane & 15);
      const float b2v = vb2S[n];
      v8f acc;
      #pragma unroll
      for (int v = 0; v < 8; ++v) acc[v] = b2v;    // bias folded into C
      #pragma unroll
      for (int kk = 0; kk < HID; kk += 32) {
        v16bf a  = load_a_frag(uS, HID, kk, lane);
        v16bf bb = load_b_frag(valW2T, HID, j * 16, kk, lane);
        acc = wmma_bf16(a, bb, acc);
      }
      const int hlf = lane >> 4;
      const float fb = fallback[n];
      #pragma unroll
      for (int v = 0; v < 8; ++v) {
        const int m = v + 8 * hlf;          // local b
        const int b = bbase + m;
        if (b < Btot)
          out[(long long)b * HID + n] = avS[m] ? acc[v] : fb;
      }
    }
  }
}

extern "C" void kernel_launch(void* const* d_in, const int* in_sizes, int n_in,
                              void* d_out, int out_size, void* d_ws, size_t ws_size,
                              hipStream_t stream) {
  const int Btot = in_sizes[0] / K_NB;
  const int numTiles = (Btot + BT - 1) / BT;
  const int grid = numTiles < MAX_PERSISTENT ? numTiles : MAX_PERSISTENT;

  const size_t smem_bf16 = (size_t)(HHALF * COMB + HID * COMB + HID * HID +
                                    ROWS * COMB + ROWS * HID + BT * HID) * sizeof(bf16_t);
  const size_t smem_f32  = (size_t)(HHALF + HHALF + HID + HID +
                                    6 * 32 + 32 + 2 * ROWS + ROWS) * sizeof(float);
  const size_t smem = smem_bf16 + smem_f32 + BT * sizeof(int);   // ~297 KB < 320 KB/WGP

  (void)hipFuncSetAttribute((const void*)sna_kernel,
                            hipFuncAttributeMaxDynamicSharedMemorySize, (int)smem);

  sna_kernel<<<grid, NTHREADS, smem, stream>>>(
      (const int*)d_in[0], (const int*)d_in[1], (const float*)d_in[2],
      (const int*)d_in[3], (const float*)d_in[4], (const unsigned char*)d_in[5],
      (const float*)d_in[6], (const float*)d_in[7], (const float*)d_in[8],
      (const float*)d_in[9], (const float*)d_in[10], (const float*)d_in[11],
      (const float*)d_in[12], (const float*)d_in[13], (const float*)d_in[14],
      (const float*)d_in[15], (const float*)d_in[16], (const float*)d_in[17],
      (const float*)d_in[18], (float*)d_out, Btot, numTiles);
}
